// GCNLayer3_79817672229558
// MI455X (gfx1250) — compile-verified, matrix operand order checked by reading, not run on hardware
//
#include <hip/hip_runtime.h>

// ---------- CDNA5 WMMA vector types ----------
typedef __attribute__((ext_vector_type(16))) __bf16         v16bf;
typedef __attribute__((ext_vector_type(8)))  float          v8f;
typedef __attribute__((ext_vector_type(8)))  unsigned short us8;

union V16 {
    v16bf          bf;
    unsigned short us[16];
    us8            h[2];
};

#define N_NODES 20000
#define D_DIM   512
#define N_EDGES 320000

// split f into bf16 hi (truncated) + bf16 lo correction; truncation keeps lo exact
__device__ __forceinline__ void bf16_split(float f, unsigned short& hi, unsigned short& lo) {
    unsigned u  = __float_as_uint(f);
    hi          = (unsigned short)(u >> 16);
    float fh    = __uint_as_float(((unsigned)hi) << 16);
    float fl    = f - fh;
    lo          = (unsigned short)(__float_as_uint(fl) >> 16);
}

// ---------- 1a) convert + transpose W -> bf16 hi/lo, column-major (Wt[n][k]) ----------
__global__ __launch_bounds__(256) void convert_w(const float* __restrict__ W,
                                                 unsigned short* __restrict__ Wt_hi,
                                                 unsigned short* __restrict__ Wt_lo) {
    int i = blockIdx.x * 256 + threadIdx.x;        // 0 .. 512*512-1
    int k = i >> 9;
    int n = i & 511;
    unsigned short hi, lo;
    bf16_split(W[i], hi, lo);
    Wt_hi[(size_t)n * D_DIM + k] = hi;
    Wt_lo[(size_t)n * D_DIM + k] = lo;
}

// ---------- 1b) convert h -> bf16 hi/lo, row-major (one-time; removes all VALU split
//               work from the GEMM hot loop) ----------
__global__ __launch_bounds__(256) void convert_h(const float* __restrict__ h,
                                                 unsigned short* __restrict__ h_hi,
                                                 unsigned short* __restrict__ h_lo) {
    size_t i = (size_t)blockIdx.x * 256 + threadIdx.x;   // 0 .. 20000*512-1
    unsigned short hi, lo;
    bf16_split(h[i], hi, lo);
    h_hi[i] = hi;
    h_lo[i] = lo;
}

// ---------- 2) zero output (atomic accumulate target) ----------
__global__ __launch_bounds__(256) void zero_out(float4* __restrict__ out) {
    out[(size_t)blockIdx.x * 256 + threadIdx.x] = make_float4(0.f, 0.f, 0.f, 0.f);
}

// ---------- 3) hw = norm * (h @ W) via bf16x3-split WMMA ----------
// One wave computes a 16x64 output slab: 1 M-tile x 4 N-tiles (4 accumulators).
// A hi/lo loaded once per k-step, reused by 12 WMMAs -> loop is b128 loads + wmma only.
// Grid exact: 1250 blocks x 8 waves; wave w owns N-tiles [4w, 4w+3]. EXEC all-ones.
__global__ __launch_bounds__(256) void gcn_gemm(const unsigned short* __restrict__ h_hi,
                                                const unsigned short* __restrict__ h_lo,
                                                const unsigned short* __restrict__ Wt_hi,
                                                const unsigned short* __restrict__ Wt_lo,
                                                const float* __restrict__ norm,
                                                float* __restrict__ hw) {
    const int lane    = threadIdx.x & 31;
    const int wave    = threadIdx.x >> 5;
    const int mt      = blockIdx.x;           // 0..1249
    const int nt_base = wave * 4;             // 0,4,...,28
    const int half    = lane >> 4;            // ISA A/B layout: lanes 16..31 hold K+8 / K+24
    const int l16     = lane & 15;
    const int m       = mt * 16 + l16;        // A row for this lane

    const unsigned short* arow_hi = h_hi + (size_t)m * D_DIM;
    const unsigned short* arow_lo = h_lo + (size_t)m * D_DIM;

    // B row base pointers for the 4 N-tiles (col = lane's l16 within each tile)
    const unsigned short* brow_hi[4];
    const unsigned short* brow_lo[4];
    #pragma unroll
    for (int j = 0; j < 4; ++j) {
        int n = (nt_base + j) * 16 + l16;
        brow_hi[j] = Wt_hi + (size_t)n * D_DIM;
        brow_lo[j] = Wt_lo + (size_t)n * D_DIM;
    }

    v8f acc0 = {}, acc1 = {}, acc2 = {}, acc3 = {};

    for (int kb = 0; kb < D_DIM; kb += 32) {
        const int k0 = kb + 8 * half;         // elements 0..7  -> K = k0 + e
        const int k1 = k0 + 16;               // elements 8..15 -> K = k1 + (e-8)

        V16 ahi, alo;
        ahi.h[0] = *(const us8*)(arow_hi + k0);
        ahi.h[1] = *(const us8*)(arow_hi + k1);
        alo.h[0] = *(const us8*)(arow_lo + k0);
        alo.h[1] = *(const us8*)(arow_lo + k1);

        #pragma unroll
        for (int j = 0; j < 4; ++j) {
            V16 bhi, blo;
            bhi.h[0] = *(const us8*)(brow_hi[j] + k0);
            bhi.h[1] = *(const us8*)(brow_hi[j] + k1);
            blo.h[0] = *(const us8*)(brow_lo[j] + k0);
            blo.h[1] = *(const us8*)(brow_lo[j] + k1);

            v8f a = (j == 0) ? acc0 : (j == 1) ? acc1 : (j == 2) ? acc2 : acc3;
            // fp32-accurate product: hi*hi + lo*hi + hi*lo
            a = __builtin_amdgcn_wmma_f32_16x16x32_bf16(false, ahi.bf, false, bhi.bf,
                                                        (short)0, a, false, false);
            a = __builtin_amdgcn_wmma_f32_16x16x32_bf16(false, alo.bf, false, bhi.bf,
                                                        (short)0, a, false, false);
            a = __builtin_amdgcn_wmma_f32_16x16x32_bf16(false, ahi.bf, false, blo.bf,
                                                        (short)0, a, false, false);
            if      (j == 0) acc0 = a;
            else if (j == 1) acc1 = a;
            else if (j == 2) acc2 = a;
            else             acc3 = a;
        }
    }

    // C/D layout: VGPR r holds row (r + 8*half), col = lane's l16 within tile
    #pragma unroll
    for (int r = 0; r < 8; ++r) {
        int   mr = mt * 16 + r + 8 * half;
        float nr = norm[mr];
        float* orow = hw + (size_t)mr * D_DIM;
        orow[(nt_base + 0) * 16 + l16] = acc0[r] * nr;
        orow[(nt_base + 1) * 16 + l16] = acc1[r] * nr;
        orow[(nt_base + 2) * 16 + l16] = acc2[r] * nr;
        orow[(nt_base + 3) * 16 + l16] = acc3[r] * nr;
    }
}

// ---------- 4) edge scatter-add: out[dst] += hw[src] ----------
// 128 threads per edge, float4 per thread, native f32 atomics
__global__ __launch_bounds__(256) void gcn_scatter(const float* __restrict__ hw,
                                                   const int* __restrict__ src,
                                                   const int* __restrict__ dst,
                                                   float* __restrict__ out) {
    unsigned gid = blockIdx.x * 256 + threadIdx.x;
    unsigned e   = gid >> 7;                  // edge id
    unsigned f   = (gid & 127) << 2;          // feature base
    int s = src[e];
    int d = dst[e];
    float4 v = *(const float4*)(hw + (size_t)s * D_DIM + f);
    float* o = out + (size_t)d * D_DIM + f;
    unsafeAtomicAdd(o + 0, v.x);
    unsafeAtomicAdd(o + 1, v.y);
    unsafeAtomicAdd(o + 2, v.z);
    unsafeAtomicAdd(o + 3, v.w);
}

// ---------- 5) out = relu(out * norm + bias), in place ----------
__global__ __launch_bounds__(256) void gcn_epilogue(float* __restrict__ out,
                                                    const float* __restrict__ norm,
                                                    const float* __restrict__ bias) {
    unsigned i   = blockIdx.x * 256 + threadIdx.x;   // float4 index
    unsigned idx = i << 2;
    unsigned row = idx >> 9;
    unsigned col = idx & 511;
    float nr = norm[row];
    float4 v = *(float4*)(out + idx);
    float4 b = *(const float4*)(bias + col);
    v.x = fmaxf(fmaf(v.x, nr, b.x), 0.f);
    v.y = fmaxf(fmaf(v.y, nr, b.y), 0.f);
    v.z = fmaxf(fmaf(v.z, nr, b.z), 0.f);
    v.w = fmaxf(fmaf(v.w, nr, b.w), 0.f);
    *(float4*)(out + idx) = v;
}

extern "C" void kernel_launch(void* const* d_in, const int* in_sizes, int n_in,
                              void* d_out, int out_size, void* d_ws, size_t ws_size,
                              hipStream_t stream) {
    const float* h    = (const float*)d_in[0];   // [20000,512]
    const float* W    = (const float*)d_in[1];   // [512,512]
    const float* bias = (const float*)d_in[2];   // [512]
    const float* norm = (const float*)d_in[3];   // [20000,1]
    const int*   src  = (const int*)d_in[4];     // [320000]
    const int*   dst  = (const int*)d_in[5];     // [320000]
    float*       out  = (float*)d_out;           // [20000,512]

    // workspace layout
    char* ws = (char*)d_ws;
    float*          hw    = (float*)ws;                                   // 20000*512 f32
    unsigned short* Wt_hi = (unsigned short*)(ws + (size_t)N_NODES * D_DIM * sizeof(float));
    unsigned short* Wt_lo = Wt_hi + (size_t)D_DIM * D_DIM;
    unsigned short* h_hi  = Wt_lo + (size_t)D_DIM * D_DIM;                // 20000*512 bf16
    unsigned short* h_lo  = h_hi + (size_t)N_NODES * D_DIM;               // 20000*512 bf16

    // precision-preserving bf16 hi/lo preprocessing (one-time, L2-resident afterwards)
    convert_w<<<(D_DIM * D_DIM) / 256, 256, 0, stream>>>(W, Wt_hi, Wt_lo);
    convert_h<<<(N_NODES * D_DIM) / 256, 256, 0, stream>>>(h, h_hi, h_lo);
    // out zero: 20000*512/4 float4s / 256
    zero_out<<<(N_NODES * D_DIM / 4) / 256, 256, 0, stream>>>((float4*)out);
    // GEMM: 1250 M-tiles, 8 waves/block, each wave does 4 N-tiles
    gcn_gemm<<<1250, 256, 0, stream>>>(h_hi, h_lo, Wt_hi, Wt_lo, norm, hw);
    // scatter: 320000 edges * 128 threads / 256
    gcn_scatter<<<(N_EDGES * 128) / 256, 256, 0, stream>>>(hw, src, dst, out);
    // epilogue
    gcn_epilogue<<<(N_NODES * D_DIM / 4) / 256, 256, 0, stream>>>(out, norm, bias);
}
